// MB_HGTBot_32031866093625
// MI455X (gfx1250) — compile-verified, matrix operand order checked by reading, not run on hardware
//
#include <hip/hip_runtime.h>
#include <hip/hip_bf16.h>

// ---------------- problem constants ----------------
#define NN     100000
#define EE     1000000
#define DD     192
#define FDIM   64
#define LLAY   6
#define TWD    768
#define TRAIN  8192
#define LRELU_SLOPE 0.01f
#define LN_EPS 1e-5f

typedef __attribute__((ext_vector_type(16))) _Float16 v16h;
typedef __attribute__((ext_vector_type(8)))  float    v8f;
typedef __attribute__((ext_vector_type(2)))  _Float16 h2;
typedef __attribute__((ext_vector_type(4)))  unsigned int u32x4;
typedef __attribute__((ext_vector_type(8)))  int      i32x8;
typedef __attribute__((ext_vector_type(4)))  int      i32x4;

// K index inside a 16x32 f16 A-fragment (ISA 7.12.2): lanes 0-15 hold
// K = {0..7,16..23}, lanes 16-31 hold K = {8..15,24..31}, packed 2 per VGPR.
__device__ __forceinline__ int kidx(int v, int hl) {
  return (v < 4) ? (2 * v + 8 * hl) : (16 + 2 * (v - 4) + 8 * hl);
}

__device__ __forceinline__ v8f wmma16(v16h a, v16h b, v8f c) {
  return __builtin_amdgcn_wmma_f32_16x16x32_f16(false, a, false, b,
                                                (short)0, c, false, false);
}

// Gather one 16x32 A fragment from an f32 row-major matrix, cvt to f16.
__device__ __forceinline__ v16h loadAfragF32(const float* __restrict__ rowbase,
                                             int kbase, int hl) {
  v16h a;
#pragma unroll
  for (int v = 0; v < 8; ++v) {
    int k = kbase + kidx(v, hl);
    float2 f = *(const float2*)(rowbase + k);
    a[2 * v]     = (_Float16)f.x;
    a[2 * v + 1] = (_Float16)f.y;
  }
  return a;
}

// Gather one fragment from an f16 row-major matrix (used for sim tiles).
__device__ __forceinline__ v16h loadFragF16(const _Float16* __restrict__ rowbase,
                                            int kbase, int hl) {
  v16h a;
#pragma unroll
  for (int v = 0; v < 8; ++v) {
    int k = kbase + kidx(v, hl);
    h2 f = *(const h2*)(rowbase + k);
    a[2 * v]     = f.x;
    a[2 * v + 1] = f.y;
  }
  return a;
}

// ---- Tensor Data Mover: DMA `nbytes` (contiguous, 8B-granular) global -> LDS.
// D# per ISA 8.3/8.4: group0 = {count=1 | lds_addr | global_addr | type=2},
// group1 = {data_size=8B, tensor_dim0=tile_dim0=nelem (1 row), tile_dim1=1}.
__device__ __forceinline__ void tdm_load_lds(_Float16* lds, const _Float16* gsrc,
                                             unsigned nbytes) {
#if __has_builtin(__builtin_amdgcn_tensor_load_to_lds)
  unsigned ldsoff = (unsigned)(unsigned long long)(void*)lds;   // flat[31:0] = LDS offset
  unsigned long long ga = (unsigned long long)(const void*)gsrc;
  unsigned nelem = nbytes >> 3;                                  // 8-byte elements
  u32x4 g0 = { 1u,                                               // count=1 user D#
               ldsoff,                                           // lds_addr
               (unsigned)(ga & 0xffffffffull),                   // global_addr[31:0]
               (unsigned)((ga >> 32) & 0x01ffffffu) | 0x80000000u }; // [56:32] | type=2
  i32x8 g1 = { (int)(3u << 16),                                  // data_size = 8B
               (int)((nelem & 0xffffu) << 16),                   // tensor_dim0[15:0]
               (int)(((nelem >> 16) & 0xffffu) | (1u << 16)),    // td0[31:16], td1=1
               (int)((nelem & 0xffffu) << 16),                   // tile_dim0 = nelem
               1,                                                // tile_dim1=1, tile_dim2=0
               (int)nelem, 0, 0 };                               // dim0 stride (unused)
  i32x4 gz = { 0, 0, 0, 0 };
#if defined(__clang_major__) && (__clang_major__ >= 23)
  i32x8 gz8 = { 0, 0, 0, 0, 0, 0, 0, 0 };
  __builtin_amdgcn_tensor_load_to_lds(g0, g1, gz, gz, gz8, 0);
#else
  __builtin_amdgcn_tensor_load_to_lds(g0, g1, gz, gz, 0);
#endif
  __builtin_amdgcn_s_wait_tensorcnt(0);
#else
  (void)lds; (void)gsrc; (void)nbytes;
#endif
}

// ---------------- weight packing: f32 (Kin x Kout) -> WMMA B fragments ------
// Fragment (kt,nt): 32 lanes x 16 halves, lane-major (32B contiguous / lane).
__global__ void pack_b_f16(const float* __restrict__ W, _Float16* __restrict__ dst,
                           int Kin, int Kout) {
  int frag = blockIdx.x;
  int lane = threadIdx.x;           // 32 threads
  int ntiles = Kout >> 4;
  int kt = frag / ntiles, nt = frag % ntiles;
  int n  = nt * 16 + (lane & 15);
  int hl = lane >> 4;
  _Float16* out = dst + (size_t)frag * 512 + lane * 16;
#pragma unroll
  for (int v = 0; v < 8; ++v) {
    int k = kt * 32 + kidx(v, hl);
    out[2 * v]     = (_Float16)W[(size_t)k * Kout + n];
    out[2 * v + 1] = (_Float16)W[(size_t)(k + 1) * Kout + n];
  }
}

// ---------------- main WMMA GEMM ----------------
// Block = 8 waves. TDM pulls the whole packed weight block (KT*NT fragments)
// into LDS once; each wave then computes 16 rows x NT*16 cols with B from LDS.
template <int NT, int KT>
__global__ void gemm_wmma(const float* __restrict__ A, int lda,
                          const _Float16* __restrict__ Bp,
                          const float* __restrict__ bias,
                          const float* __restrict__ residual,
                          float* __restrict__ out, int ldc, int col0,
                          int ntilerows, int act) {
  extern __shared__ _Float16 smem[];
  constexpr unsigned WBYTES = (unsigned)NT * KT * 512u * 2u;
#if __has_builtin(__builtin_amdgcn_tensor_load_to_lds)
  if (threadIdx.x < 32) tdm_load_lds(smem, Bp, WBYTES);
#else
  for (unsigned i = threadIdx.x; i < WBYTES / 8u; i += blockDim.x)
    ((unsigned long long*)smem)[i] = ((const unsigned long long*)Bp)[i];
#endif
  __syncthreads();

  int wave = threadIdx.x >> 5;
  int lane = threadIdx.x & 31;
  int tile = blockIdx.x * (blockDim.x >> 5) + wave;
  if (tile >= ntilerows) return;     // after the barrier: no divergence hazard
  int row0 = tile * 16;
  int m  = lane & 15;
  int hl = lane >> 4;
  const float* Arow = A + (size_t)(row0 + m) * lda;

  v8f acc[NT];
#pragma unroll
  for (int nt = 0; nt < NT; ++nt)
#pragma unroll
    for (int r = 0; r < 8; ++r) acc[nt][r] = 0.0f;

  for (int kt = 0; kt < KT; ++kt) {
    if (kt + 1 < KT) __builtin_prefetch(Arow + (kt + 1) * 32, 0, 1);
    v16h a = loadAfragF32(Arow, kt * 32, hl);
    const _Float16* bb = smem + (size_t)(kt * NT) * 512 + lane * 16;
#pragma unroll
    for (int nt = 0; nt < NT; ++nt) {
      v16h b = *(const v16h*)(bb + (size_t)nt * 512);
      acc[nt] = wmma16(a, b, acc[nt]);
    }
  }

  int rbase = row0 + 8 * hl;   // C/D layout: VGPR r -> row r + 8*hl, lane&15 -> col
#pragma unroll
  for (int nt = 0; nt < NT; ++nt) {
    float bval = bias ? bias[nt * 16 + m] : 0.0f;
#pragma unroll
    for (int r = 0; r < 8; ++r) {
      float v = acc[nt][r] + bval;
      if (act) v = (v > 0.0f) ? v : LRELU_SLOPE * v;
      size_t idx = (size_t)(rbase + r) * ldc + col0 + nt * 16 + m;
      if (residual) v += residual[idx];
      out[idx] = v;
    }
  }
}

// ---------------- small dense kernels ----------------
__global__ void kzero(float* __restrict__ p, size_t n) {
  size_t i = (size_t)blockIdx.x * blockDim.x + threadIdx.x;
  if (i < n) p[i] = 0.0f;
}

// lrelu(num_prop@Wn + bn) -> x[:,64:128], lrelu(comments@Wc + bc) -> x[:,128:192]
__global__ void small_proj(const float* __restrict__ np, const float* __restrict__ cp,
                           const float* __restrict__ Wn, const float* __restrict__ bn_,
                           const float* __restrict__ Wc, const float* __restrict__ bc,
                           float* __restrict__ x) {
  int i = blockIdx.x * blockDim.x + threadIdx.x;
  if (i >= NN) return;
  float a6[6], a7[7];
#pragma unroll
  for (int k = 0; k < 6; ++k) a6[k] = np[(size_t)i * 6 + k];
#pragma unroll
  for (int k = 0; k < 7; ++k) a7[k] = cp[(size_t)i * 7 + k];
  float* xr = x + (size_t)i * DD;
  for (int o = 0; o < FDIM; ++o) {
    float s = bn_[o];
#pragma unroll
    for (int k = 0; k < 6; ++k) s += a6[k] * Wn[k * FDIM + o];
    xr[FDIM + o] = (s > 0.0f) ? s : LRELU_SLOPE * s;
    float c = bc[o];
#pragma unroll
    for (int k = 0; k < 7; ++k) c += a7[k] * Wc[k * FDIM + o];
    xr[2 * FDIM + o] = (c > 0.0f) ? c : LRELU_SLOPE * c;
  }
}

// LayerNorm over last dim (192). One wave per row.
__global__ void ln_kernel(float* __restrict__ x, const float* __restrict__ g,
                          const float* __restrict__ b, int nrows) {
  int wave = threadIdx.x >> 5, lane = threadIdx.x & 31;
  int row = blockIdx.x * (blockDim.x >> 5) + wave;
  if (row >= nrows) return;
  float* xr = x + (size_t)row * DD;
  float v[6];
  float s = 0.0f;
#pragma unroll
  for (int j = 0; j < 6; ++j) { v[j] = xr[lane + 32 * j]; s += v[j]; }
#pragma unroll
  for (int msk = 1; msk < 32; msk <<= 1) s += __shfl_xor(s, msk, 32);
  float mu = s * (1.0f / DD);
  float q = 0.0f;
#pragma unroll
  for (int j = 0; j < 6; ++j) { float d = v[j] - mu; q += d * d; }
#pragma unroll
  for (int msk = 1; msk < 32; msk <<= 1) q += __shfl_xor(q, msk, 32);
  float rstd = rsqrtf(q * (1.0f / DD) + LN_EPS);
#pragma unroll
  for (int j = 0; j < 6; ++j) {
    int d = lane + 32 * j;
    xr[d] = (v[j] - mu) * rstd * g[d] + b[d];
  }
}

// BatchNorm over rows: stage 1 per-column sum / sumsq, stage 2 apply.
__global__ void bn_reduce(const float* __restrict__ x, float* __restrict__ sums,
                          float* __restrict__ sqs) {
  int d = blockIdx.x;                 // 0..191
  int r0 = blockIdx.y * 3125;
  int r1 = r0 + 3125; if (r1 > NN) r1 = NN;
  float s = 0.0f, q = 0.0f;
  for (int i = r0 + threadIdx.x; i < r1; i += blockDim.x) {
    float v = x[(size_t)i * DD + d];
    s += v; q += v * v;
  }
  __shared__ float ss[256], qq[256];
  ss[threadIdx.x] = s; qq[threadIdx.x] = q;
  __syncthreads();
  for (int k = 128; k > 0; k >>= 1) {
    if (threadIdx.x < k) { ss[threadIdx.x] += ss[threadIdx.x + k]; qq[threadIdx.x] += qq[threadIdx.x + k]; }
    __syncthreads();
  }
  if (threadIdx.x == 0) { atomicAdd(&sums[d], ss[0]); atomicAdd(&sqs[d], qq[0]); }
}

__global__ void bn_apply(float* __restrict__ x, const float* __restrict__ sums,
                         const float* __restrict__ sqs, const float* __restrict__ g,
                         const float* __restrict__ b) {
  size_t idx = (size_t)blockIdx.x * blockDim.x + threadIdx.x;
  if (idx >= (size_t)NN * DD) return;
  int d = (int)(idx % DD);
  float mu  = sums[d] * (1.0f / NN);
  float var = sqs[d] * (1.0f / NN) - mu * mu;
  x[idx] = (x[idx] - mu) * rsqrtf(var + LN_EPS) * g[d] + b[d];
}

// Row L2-normalize first TRAIN rows -> f16 matrix for WMMA sim tiles.
__global__ void rownorm_f16(const float* __restrict__ x, _Float16* __restrict__ fn) {
  int wave = threadIdx.x >> 5, lane = threadIdx.x & 31;
  int row = blockIdx.x * (blockDim.x >> 5) + wave;
  if (row >= TRAIN) return;
  const float* xr = x + (size_t)row * DD;
  float v[6];
  float s = 0.0f;
#pragma unroll
  for (int j = 0; j < 6; ++j) { v[j] = xr[lane + 32 * j]; s += v[j] * v[j]; }
#pragma unroll
  for (int msk = 1; msk < 32; msk <<= 1) s += __shfl_xor(s, msk, 32);
  float rn = rsqrtf(s);
#pragma unroll
  for (int j = 0; j < 6; ++j) fn[(size_t)row * DD + lane + 32 * j] = (_Float16)(v[j] * rn);
}

// Contrastive-loss tiles: wave = 16 rows x 512 cols (32 WMMA tiles); sim never
// hits memory. exp(sim/TEMP) = exp(2*dot). B frags preloaded (6 in flight) so
// the WMMA chain is not gated on single loads.
__global__ void sim_tile(const _Float16* __restrict__ fn, const int* __restrict__ labels,
                         float* __restrict__ acc_match, float* __restrict__ acc_ex) {
  int lane = threadIdx.x;             // block = 32 threads = 1 wave
  int row0 = blockIdx.x * 16;
  int m  = lane & 15;
  int hl = lane >> 4;
  const _Float16* arow = fn + (size_t)(row0 + m) * DD;
  v16h afr[6];
#pragma unroll
  for (int kt = 0; kt < 6; ++kt) afr[kt] = loadFragF16(arow, kt * 32, hl);
  int labrow[8];
#pragma unroll
  for (int r = 0; r < 8; ++r) labrow[r] = labels[row0 + r + 8 * hl];

  float sume[8], summ[8];
#pragma unroll
  for (int r = 0; r < 8; ++r) { sume[r] = 0.0f; summ[r] = 0.0f; }

  int ct0 = blockIdx.y * 32;          // 16 col-chunks x 32 tiles = 512 tiles
  for (int ct = ct0; ct < ct0 + 32; ++ct) {
    int col0 = ct * 16;
    const _Float16* brow = fn + (size_t)(col0 + m) * DD;
    int labcol = labels[col0 + m];
    v16h bfr[6];
#pragma unroll
    for (int kt = 0; kt < 6; ++kt) bfr[kt] = loadFragF16(brow, kt * 32, hl);
    v8f acc;
#pragma unroll
    for (int r = 0; r < 8; ++r) acc[r] = 0.0f;
#pragma unroll
    for (int kt = 0; kt < 6; ++kt) acc = wmma16(afr[kt], bfr[kt], acc);
#pragma unroll
    for (int r = 0; r < 8; ++r) {
      float e = __expf(2.0f * acc[r]);     // /TEMP with TEMP=0.5
      sume[r] += e;
      summ[r] += (labrow[r] == labcol) ? e : 0.0f;
    }
  }
#pragma unroll
  for (int r = 0; r < 8; ++r) {
    float se = sume[r], sm = summ[r];
#pragma unroll
    for (int msk = 1; msk < 16; msk <<= 1) {       // reduce 16-lane half-group
      se += __shfl_xor(se, msk, 32);
      sm += __shfl_xor(sm, msk, 32);
    }
    if (m == 0) {
      atomicAdd(&acc_ex[row0 + r + 8 * hl], se);
      atomicAdd(&acc_match[row0 + r + 8 * hl], sm);
    }
  }
}

__global__ void closs_final(const float* __restrict__ am, const float* __restrict__ ae,
                            float* __restrict__ out) {
  __shared__ float red[256];
  int tid = threadIdx.x;
  float s = 0.0f;
  for (int i = tid; i < TRAIN; i += 256) s += -__logf(am[i] / ae[i]);
  red[tid] = s; __syncthreads();
  for (int k = 128; k > 0; k >>= 1) {
    if (tid < k) red[tid] += red[tid + k];
    __syncthreads();
  }
  if (tid == 0) out[0] = red[0] * (1.0f / TRAIN);
}

// ---------------- RGCN edge kernels ----------------
__global__ void edge_count(const int* __restrict__ ei, const int* __restrict__ et,
                           float* __restrict__ cnt) {
  int e = blockIdx.x * blockDim.x + threadIdx.x;
  if (e >= EE) return;
  atomicAdd(&cnt[(size_t)et[e] * NN + ei[EE + e]], 1.0f);
}

__global__ void edge_scatter(const int* __restrict__ ei, const int* __restrict__ et,
                             const float* __restrict__ h, float* __restrict__ agg,
                             int rel) {
  int e = blockIdx.x;
  if (et[e] != rel) return;
  int s = ei[e], t = ei[EE + e];
  int d = threadIdx.x;                // 192 lanes, coalesced
  atomicAdd(&agg[(size_t)t * DD + d], h[(size_t)s * DD + d]);
}

__global__ void rgcn_combine(float* __restrict__ out, const float* __restrict__ agg0,
                             const float* __restrict__ agg1, const float* __restrict__ cnt) {
  size_t idx = (size_t)blockIdx.x * blockDim.x + threadIdx.x;
  if (idx >= (size_t)NN * DD) return;
  int i = (int)(idx / DD);
  out[idx] += agg0[idx] / fmaxf(cnt[i], 1.0f) + agg1[idx] / fmaxf(cnt[NN + i], 1.0f);
}

// Final 192 -> 2 head (too thin for WMMA).
__global__ void logits_k(const float* __restrict__ h, const float* __restrict__ Wm3,
                         const float* __restrict__ bm3, float* __restrict__ out) {
  int i = blockIdx.x * blockDim.x + threadIdx.x;
  if (i >= NN) return;
  float s0 = bm3[0], s1 = bm3[1];
  const float* hr = h + (size_t)i * DD;
  for (int d = 0; d < DD; ++d) {
    float v = hr[d];
    s0 += v * Wm3[2 * d];
    s1 += v * Wm3[2 * d + 1];
  }
  out[2 * i] = s0;
  out[2 * i + 1] = s1;
}

// ---------------- orchestration ----------------
extern "C" void kernel_launch(void* const* d_in, const int* in_sizes, int n_in,
                              void* d_out, int out_size, void* d_ws, size_t ws_size,
                              hipStream_t stream) {
  const float* tweet = (const float*)d_in[0];
  const float* nump  = (const float*)d_in[1];
  const float* comp  = (const float*)d_in[2];
  const float* Wt = (const float*)d_in[3];  const float* bt  = (const float*)d_in[4];
  const float* Wn = (const float*)d_in[5];  const float* bnm = (const float*)d_in[6];
  const float* Wc = (const float*)d_in[7];  const float* bc  = (const float*)d_in[8];
  const float* Wv = (const float*)d_in[9];  const float* bv  = (const float*)d_in[10];
  const float* Wo = (const float*)d_in[11]; const float* bo  = (const float*)d_in[12];
  const float* ln1g = (const float*)d_in[13]; const float* ln1b = (const float*)d_in[14];
  const float* bng  = (const float*)d_in[15]; const float* bnb  = (const float*)d_in[16];
  const float* W1 = (const float*)d_in[17]; const float* b1 = (const float*)d_in[18];
  const float* W2 = (const float*)d_in[19]; const float* b2 = (const float*)d_in[20];
  const float* ln2g = (const float*)d_in[21]; const float* ln2b = (const float*)d_in[22];
  const float* Wrel = (const float*)d_in[23];
  const float* Wroot = (const float*)d_in[24]; const float* rgb = (const float*)d_in[25];
  const float* Wm1 = (const float*)d_in[26]; const float* bm1 = (const float*)d_in[27];
  const float* Wm2 = (const float*)d_in[28]; const float* bm2 = (const float*)d_in[29];
  const float* Wm3 = (const float*)d_in[30]; const float* bm3 = (const float*)d_in[31];
  const int* edge_index = (const int*)d_in[32];
  const int* edge_type  = (const int*)d_in[33];
  const int* labels     = (const int*)d_in[34];
  float* out = (float*)d_out;

  // ---- workspace layout ----
  char* ws = (char*)d_ws;
  size_t off = 0;
  auto take = [&](size_t bytes) -> char* {
    char* p = ws + off;
    off = (off + bytes + 255) & ~(size_t)255;
    return p;
  };
  const size_t ND = (size_t)NN * DD;
  float* xbuf = (float*)take(ND * 4);
  float* ybuf = (float*)take(ND * 4);
  float* zbuf = (float*)take(ND * 4);
  float* agg0 = (float*)take(ND * 4);
  float* agg1 = (float*)take(ND * 4);
  float* cnt  = (float*)take((size_t)2 * NN * 4);
  float* bnstats = (float*)take(2 * DD * 4);      // [0..191]=sum, [192..383]=sumsq
  float* racm = (float*)take(TRAIN * 4);
  float* race = (float*)take(TRAIN * 4);
  _Float16* fn16 = (_Float16*)take((size_t)TRAIN * DD * 2);
  _Float16* wp   = (_Float16*)take((size_t)1118208 * 2);   // packed f16 weights
  (void)ws_size; (void)n_in; (void)in_sizes; (void)out_size;

  // packed weight sub-offsets (in halves)
  const size_t OFF_WT = 0;                               // 24*4*512
  const size_t SZ192  = 72 * 512;                        // one 192x192 matrix
  const size_t OFF_WV = 49152;
  const size_t OFF_WO = OFF_WV + 6 * SZ192;
  const size_t OFF_W1 = OFF_WO + 6 * SZ192;
  const size_t OFF_W2 = OFF_W1 + 6 * SZ192;
  const size_t OFF_R0 = OFF_W2 + 6 * SZ192;
  const size_t OFF_R1 = OFF_R0 + SZ192;
  const size_t OFF_RT = OFF_R1 + SZ192;
  const size_t OFF_M1 = OFF_RT + SZ192;
  const size_t OFF_M2 = OFF_M1 + SZ192;

  auto packm = [&](const float* W, size_t dsto, int Kin, int Kout) {
    int frags = (Kin / 32) * (Kout / 16);
    pack_b_f16<<<frags, 32, 0, stream>>>(W, wp + dsto, Kin, Kout);
  };
  packm(Wt, OFF_WT, TWD, FDIM);
  for (int i = 0; i < LLAY; ++i) {
    packm(Wv + (size_t)i * DD * DD, OFF_WV + i * SZ192, DD, DD);
    packm(Wo + (size_t)i * DD * DD, OFF_WO + i * SZ192, DD, DD);
    packm(W1 + (size_t)i * DD * DD, OFF_W1 + i * SZ192, DD, DD);
    packm(W2 + (size_t)i * DD * DD, OFF_W2 + i * SZ192, DD, DD);
  }
  packm(Wrel,                 OFF_R0, DD, DD);
  packm(Wrel + (size_t)DD*DD, OFF_R1, DD, DD);
  packm(Wroot, OFF_RT, DD, DD);
  packm(Wm1,   OFF_M1, DD, DD);
  packm(Wm2,   OFF_M2, DD, DD);

  const int RTILES = NN / 16;                        // 6250
  const int GBLK   = (RTILES + 7) / 8;               // 8 waves / block
  const unsigned LDS12 = 12 * 6 * 512 * 2;           // 72 KB packed weights
  const unsigned LDS4  = 4 * 24 * 512 * 2;           // 96 KB (tweet proj)
  auto gemm12 = [&](const float* A, size_t bo_, const float* bias,
                    const float* res, float* o, int act) {
    gemm_wmma<12, 6><<<GBLK, 256, LDS12, stream>>>(A, DD, wp + bo_, bias, res,
                                                   o, DD, 0, RTILES, act);
  };
  auto zero = [&](float* p, size_t n) {
    kzero<<<(unsigned)((n + 255) / 256), 256, 0, stream>>>(p, n);
  };

  // ---- input projection: x = [lrelu(tweet@Wt), lrelu(np@Wn), lrelu(cp@Wc)] ----
  gemm_wmma<4, 24><<<GBLK, 256, LDS4, stream>>>(tweet, TWD, wp + OFF_WT, bt,
                                                nullptr, xbuf, DD, 0, RTILES, 1);
  small_proj<<<(NN + 255) / 256, 256, 0, stream>>>(nump, comp, Wn, bnm, Wc, bc, xbuf);

  // ---- 6 transformer-style layers ----
  const int LNG = (NN + 7) / 8;
  for (int i = 0; i < LLAY; ++i) {
    gemm12(xbuf, OFF_WV + i * SZ192, bv + i * DD, nullptr, ybuf, 0);       // v
    gemm12(ybuf, OFF_WO + i * SZ192, bo + i * DD, xbuf, xbuf, 0);          // x += v@Wo+bo
    ln_kernel<<<LNG, 256, 0, stream>>>(xbuf, ln1g + i * DD, ln1b + i * DD, NN);
    zero(bnstats, 2 * DD);
    bn_reduce<<<dim3(DD, 32), 256, 0, stream>>>(xbuf, bnstats, bnstats + DD);
    bn_apply<<<(unsigned)((ND + 255) / 256), 256, 0, stream>>>(xbuf, bnstats,
                                        bnstats + DD, bng + i * DD, bnb + i * DD);
    gemm12(xbuf, OFF_W1 + i * SZ192, b1 + i * DD, nullptr, ybuf, 1);       // lrelu(x@W1)
    gemm12(ybuf, OFF_W2 + i * SZ192, b2 + i * DD, xbuf, xbuf, 0);          // x += f
    ln_kernel<<<LNG, 256, 0, stream>>>(xbuf, ln2g + i * DD, ln2b + i * DD, NN);
  }

  // ---- contrastive loss on first 8192 rows (before RGCN mutates buffers) ----
  rownorm_f16<<<TRAIN / 8, 256, 0, stream>>>(xbuf, fn16);
  zero(racm, TRAIN); zero(race, TRAIN);
  sim_tile<<<dim3(TRAIN / 16, 16), 32, 0, stream>>>(fn16, labels, racm, race);
  closs_final<<<1, 256, 0, stream>>>(racm, race, out + (size_t)2 * NN);

  // ---- RGCN x2 ----
  zero(cnt, (size_t)2 * NN);
  edge_count<<<(EE + 255) / 256, 256, 0, stream>>>(edge_index, edge_type, cnt);

  auto rgcn_layer = [&](const float* in, float* o) {
    zero(agg0, ND); zero(agg1, ND);
    gemm12(in, OFF_R0, nullptr, nullptr, ybuf, 0);
    edge_scatter<<<EE, DD, 0, stream>>>(edge_index, edge_type, ybuf, agg0, 0);
    gemm12(in, OFF_R1, nullptr, nullptr, ybuf, 0);
    edge_scatter<<<EE, DD, 0, stream>>>(edge_index, edge_type, ybuf, agg1, 1);
    gemm12(in, OFF_RT, rgb, nullptr, o, 0);
    rgcn_combine<<<(unsigned)((ND + 255) / 256), 256, 0, stream>>>(o, agg0, agg1, cnt);
  };
  rgcn_layer(xbuf, zbuf);
  rgcn_layer(zbuf, xbuf);

  // ---- output MLP ----
  gemm12(xbuf, OFF_M1, bm1, nullptr, ybuf, 1);
  gemm12(ybuf, OFF_M2, bm2, nullptr, zbuf, 1);
  logits_k<<<(NN + 255) / 256, 256, 0, stream>>>(zbuf, Wm3, bm3, out);
}
